// GeneSetEncoder_68324339744810
// MI455X (gfx1250) — matrix operand order repaired and checked by reading, not compiled, not run on hardware
//
#include <hip/hip_runtime.h>
#include <math.h>

typedef __attribute__((ext_vector_type(16))) __bf16 v16bf;
typedef __attribute__((ext_vector_type(8)))  float  v8f;
typedef __attribute__((ext_vector_type(4)))  unsigned int u32x4;

#define NG   4096
#define NB   64
#define NS   50
#define NL   256
#define ND   128
#define NH   4
#define NDH  32

// ---- helpers -------------------------------------------------------------

__device__ __forceinline__ v16bf ldfrag(const __bf16* p0, const __bf16* p1) {
  union { u32x4 u[2]; v16bf v; } c;
  c.u[0] = *(const u32x4*)p0;
  c.u[1] = *(const u32x4*)p1;
  return c.v;
}

__device__ __forceinline__ v8f vzero() {
  v8f z = {0.f, 0.f, 0.f, 0.f, 0.f, 0.f, 0.f, 0.f};
  return z;
}

__device__ __forceinline__ v8f wmma_bf16(v16bf a, v16bf b, v8f c) {
  return __builtin_amdgcn_wmma_f32_16x16x32_bf16(false, a, false, b,
                                                 (short)0, c, false, false);
}

// ---- weight fp32 -> bf16 -------------------------------------------------

__global__ void gse_cvt(const float* __restrict__ a, __bf16* __restrict__ o, int n) {
  int i = blockIdx.x * 256 + threadIdx.x;
  if (i < n) o[i] = (__bf16)a[i];
}

// ---- fused embed-scale + proj GEMM + LN + GeLU --------------------------
// rows = NB*NG, each wave computes a 16x128 tile (4 K-steps of wmma bf16)

__global__ __launch_bounds__(256) void gse_proj(
    const float* __restrict__ expr, const float* __restrict__ embed,
    const __bf16* __restrict__ wb, const float* __restrict__ bias,
    const float* __restrict__ lng, const float* __restrict__ lnb,
    __bf16* __restrict__ xout)
{
  const int lane = threadIdx.x & 31;
  const int wave = threadIdx.x >> 5;
  const int half = lane >> 4;
  const int ln15 = lane & 15;
  const int mt   = blockIdx.x * 8 + wave;

  // A-matrix row for this lane
  const int arow = mt * 16 + ln15;
  const int b    = arow >> 12;          // / NG
  const int gene = arow & (NG - 1);
  const float ex = expr[b * NG + gene];
  const float* er = embed + (size_t)gene * ND;

  v16bf af[4];
  #pragma unroll
  for (int kb = 0; kb < 4; ++kb) {
    const float* p0 = er + kb * 32 + 16 * half;  // k = kb*32 + 16*half + e
    const float* p1 = p0 + 8;                    // k = kb*32 + 16*half + 8 + e
    v16bf a;
    #pragma unroll
    for (int e = 0; e < 8; ++e) {
      a[e]     = (__bf16)(p0[e] * ex);
      a[e + 8] = (__bf16)(p1[e] * ex);
    }
    af[kb] = a;
  }

  v8f acc[8];
  #pragma unroll
  for (int nt = 0; nt < 8; ++nt) {
    v8f c = vzero();
    const int col = nt * 16 + ln15;
    #pragma unroll
    for (int kb = 0; kb < 4; ++kb) {
      const __bf16* wp = wb + (size_t)col * ND + kb * 32 + 16 * half;
      c = wmma_bf16(af[kb], ldfrag(wp, wp + 8), c);
    }
    const float bv = bias[col];
    #pragma unroll
    for (int v = 0; v < 8; ++v) c[v] += bv;
    acc[nt] = c;
  }

  float gv[8], bv2[8];
  #pragma unroll
  for (int nt = 0; nt < 8; ++nt) {
    gv[nt]  = lng[nt * 16 + ln15];
    bv2[nt] = lnb[nt * 16 + ln15];
  }

  // LayerNorm per row (row m = 8*half + v lives in the 16 lanes of a half)
  #pragma unroll
  for (int v = 0; v < 8; ++v) {
    float sm = 0.f, s2 = 0.f;
    #pragma unroll
    for (int nt = 0; nt < 8; ++nt) { float xx = acc[nt][v]; sm += xx; s2 += xx * xx; }
    for (int m = 1; m <= 8; m <<= 1) {
      sm += __shfl_xor(sm, m, 32);
      s2 += __shfl_xor(s2, m, 32);
    }
    const float mu  = sm * (1.0f / 128.0f);
    const float var = fmaxf(s2 * (1.0f / 128.0f) - mu * mu, 0.f);
    const float rs  = rsqrtf(var + 1e-5f);
    __bf16* orow = xout + (size_t)(mt * 16 + 8 * half + v) * ND;
    #pragma unroll
    for (int nt = 0; nt < 8; ++nt) {
      const float y  = (acc[nt][v] - mu) * rs * gv[nt] + bv2[nt];
      const float ge = 0.5f * y * (1.0f + erff(y * 0.70710678118f));
      orow[nt * 16 + ln15] = (__bf16)ge;
    }
  }
}

// ---- per (batch,set): gather + QKV + flash attention + out-proj + LN + mean

__global__ __launch_bounds__(256) void gse_attn(
    const __bf16* __restrict__ x, const int* __restrict__ sidx,
    const unsigned char* __restrict__ svalid,
    const __bf16* __restrict__ inw, const float* __restrict__ inb,
    const __bf16* __restrict__ outw, const float* __restrict__ outb,
    const float* __restrict__ ln2g, const float* __restrict__ ln2b,
    float* __restrict__ out)
{
  __shared__ __bf16 qo[NL * ND];          // q, later overwritten by o (64KB)
  __shared__ __bf16 kbf[NL * ND];         // k row-major                (64KB)
  __shared__ __bf16 vt[ND * NL];          // v transposed [d][m]        (64KB)
  __shared__ __bf16 pb[8 * 16 * 32];      // per-wave P bounce          (8KB)
  __shared__ int    idxs[NL];
  __shared__ float  vmk[NL];
  __shared__ float  oacc[ND];
  __shared__ float  cnt;

  const int tid  = threadIdx.x;
  const int lane = tid & 31, wave = tid >> 5;
  const int half = lane >> 4, ln15 = lane & 15;
  const int b = blockIdx.x, s = blockIdx.y;

  if (tid < NL) {
    idxs[tid] = sidx[s * NL + tid];
    vmk[tid]  = svalid[s * NL + tid] ? 1.0f : 0.0f;
  }
  if (tid < ND) oacc[tid] = 0.f;
  __syncthreads();
  if (tid == 0) {
    float c = 0.f;
    for (int i = 0; i < NL; ++i) c += vmk[i];
    cnt = c;
  }

  // ---------------- Phase 1: QKV = g @ inW^T + b ----------------
  for (int mt = wave * 2; mt < wave * 2 + 2; ++mt) {
    const int rw = mt * 16 + ln15;
    const __bf16* gr = x + ((size_t)b * NG + idxs[rw]) * ND;
    v16bf af[4];
    #pragma unroll
    for (int kb = 0; kb < 4; ++kb) {
      const __bf16* p0 = gr + kb * 32 + 16 * half;
      af[kb] = ldfrag(p0, p0 + 8);
    }
    for (int nt = 0; nt < 24; ++nt) {
      v8f c = vzero();
      const int col = nt * 16 + ln15;
      #pragma unroll
      for (int kb = 0; kb < 4; ++kb) {
        const __bf16* wp = inw + (size_t)col * ND + kb * 32 + 16 * half;
        c = wmma_bf16(af[kb], ldfrag(wp, wp + 8), c);
      }
      const float bias = inb[col];
      const int m0 = mt * 16 + 8 * half;
      #pragma unroll
      for (int v = 0; v < 8; ++v) {
        const __bf16 val = (__bf16)(c[v] + bias);
        if (nt < 8)       qo[(m0 + v) * ND + nt * 16 + ln15] = val;
        else if (nt < 16) kbf[(m0 + v) * ND + (nt - 8) * 16 + ln15] = val;
        else              vt[((nt - 16) * 16 + ln15) * NL + (m0 + v)] = val;
      }
    }
  }
  __syncthreads();

  // ---------------- Phase 2: attention (flash-style) ----------------
  const float scl = 0.17677669529663687f;   // 1/sqrt(32)
  __bf16* pw = pb + wave * (16 * 32);
  for (int i = 0; i < 8; ++i) {
    const int t = wave + 8 * i;
    const int h = t >> 4, qt = t & 15;
    const __bf16* qr = qo + (size_t)(qt * 16 + ln15) * ND + h * 32 + 16 * half;
    const v16bf aq = ldfrag(qr, qr + 8);
    v8f o0 = vzero(), o1 = vzero();
    float rmax[8], rsum[8];
    #pragma unroll
    for (int v = 0; v < 8; ++v) { rmax[v] = -1e30f; rsum[v] = 0.f; }

    for (int kc = 0; kc < 8; ++kc) {
      const int key0 = kc * 32 + ln15;
      const __bf16* kp0 = kbf + (size_t)key0 * ND + h * 32 + 16 * half;
      const __bf16* kp1 = kbf + (size_t)(key0 + 16) * ND + h * 32 + 16 * half;
      v8f s0 = wmma_bf16(aq, ldfrag(kp0, kp0 + 8), vzero());
      v8f s1 = wmma_bf16(aq, ldfrag(kp1, kp1 + 8), vzero());
      const bool inv0 = vmk[key0] < 0.5f;
      const bool inv1 = vmk[key0 + 16] < 0.5f;
      #pragma unroll
      for (int v = 0; v < 8; ++v) {
        s0[v] = inv0 ? -1e30f : s0[v] * scl;
        s1[v] = inv1 ? -1e30f : s1[v] * scl;
      }
      #pragma unroll
      for (int v = 0; v < 8; ++v) {
        float mx = fmaxf(s0[v], s1[v]);
        for (int m = 1; m <= 8; m <<= 1) mx = fmaxf(mx, __shfl_xor(mx, m, 32));
        const float mn = fmaxf(rmax[v], mx);
        const float fac = __expf(rmax[v] - mn);
        rmax[v] = mn;
        const float p0 = __expf(s0[v] - mn);
        const float p1 = __expf(s1[v] - mn);
        float rsp = p0 + p1;
        for (int m = 1; m <= 8; m <<= 1) rsp += __shfl_xor(rsp, m, 32);
        rsum[v] = rsum[v] * fac + rsp;
        o0[v] *= fac;
        o1[v] *= fac;
        pw[(8 * half + v) * 32 + ln15]      = (__bf16)p0;
        pw[(8 * half + v) * 32 + 16 + ln15] = (__bf16)p1;
      }
      const __bf16* pr = pw + ln15 * 32 + 16 * half;     // A-frag of P(16x32)
      const v16bf ap = ldfrag(pr, pr + 8);
      const __bf16* vp0 = vt + (size_t)(h * 32 + ln15) * NL + kc * 32 + 16 * half;
      const __bf16* vp1 = vt + (size_t)(h * 32 + 16 + ln15) * NL + kc * 32 + 16 * half;
      o0 = wmma_bf16(ap, ldfrag(vp0, vp0 + 8), o0);
      o1 = wmma_bf16(ap, ldfrag(vp1, vp1 + 8), o1);
    }

    const int m0 = qt * 16 + 8 * half;
    #pragma unroll
    for (int v = 0; v < 8; ++v) {
      const float inv = 1.0f / fmaxf(rsum[v], 1e-30f);
      qo[(m0 + v) * ND + h * 32 + ln15]      = (__bf16)(o0[v] * inv);
      qo[(m0 + v) * ND + h * 32 + 16 + ln15] = (__bf16)(o1[v] * inv);
    }
  }
  __syncthreads();

  // ---------------- Phase 3: out-proj + residual LN + masked mean ----------
  float lgv[8], lbv[8];
  #pragma unroll
  for (int nt = 0; nt < 8; ++nt) {
    lgv[nt] = ln2g[nt * 16 + ln15];
    lbv[nt] = ln2b[nt * 16 + ln15];
  }
  for (int mt = wave * 2; mt < wave * 2 + 2; ++mt) {
    const __bf16* orow = qo + (size_t)(mt * 16 + ln15) * ND;
    v16bf af[4];
    #pragma unroll
    for (int kb = 0; kb < 4; ++kb) {
      const __bf16* p0 = orow + kb * 32 + 16 * half;
      af[kb] = ldfrag(p0, p0 + 8);
    }
    v8f acc[8];
    const int m0 = mt * 16 + 8 * half;
    #pragma unroll
    for (int nt = 0; nt < 8; ++nt) {
      v8f c = vzero();
      const int col = nt * 16 + ln15;
      #pragma unroll
      for (int kb = 0; kb < 4; ++kb) {
        const __bf16* wp = outw + (size_t)col * ND + kb * 32 + 16 * half;
        c = wmma_bf16(af[kb], ldfrag(wp, wp + 8), c);
      }
      const float bias = outb[col];
      #pragma unroll
      for (int v = 0; v < 8; ++v) {
        const __bf16 g = x[((size_t)b * NG + idxs[m0 + v]) * ND + col];
        c[v] += bias + (float)g;           // residual g + o
      }
      acc[nt] = c;
    }
    float colsum[8];
    #pragma unroll
    for (int nt = 0; nt < 8; ++nt) colsum[nt] = 0.f;
    #pragma unroll
    for (int v = 0; v < 8; ++v) {
      float sm = 0.f, s2 = 0.f;
      #pragma unroll
      for (int nt = 0; nt < 8; ++nt) { float xx = acc[nt][v]; sm += xx; s2 += xx * xx; }
      for (int m = 1; m <= 8; m <<= 1) {
        sm += __shfl_xor(sm, m, 32);
        s2 += __shfl_xor(s2, m, 32);
      }
      const float mu  = sm * (1.0f / 128.0f);
      const float var = fmaxf(s2 * (1.0f / 128.0f) - mu * mu, 0.f);
      const float rs  = rsqrtf(var + 1e-5f);
      const float vf  = vmk[m0 + v];
      #pragma unroll
      for (int nt = 0; nt < 8; ++nt) {
        const float y = (acc[nt][v] - mu) * rs * lgv[nt] + lbv[nt];
        colsum[nt] += y * vf;
      }
    }
    #pragma unroll
    for (int nt = 0; nt < 8; ++nt) atomicAdd(&oacc[nt * 16 + ln15], colsum[nt]);
  }
  __syncthreads();
  if (tid < ND) out[((size_t)b * NS + s) * ND + tid] = oacc[tid] / fmaxf(cnt, 1.0f);
}

// ---- launcher ------------------------------------------------------------

extern "C" void kernel_launch(void* const* d_in, const int* in_sizes, int n_in,
                              void* d_out, int out_size, void* d_ws, size_t ws_size,
                              hipStream_t stream) {
  const float* gene_expr   = (const float*)d_in[0];
  const int*   set_indices = (const int*)d_in[1];
  const unsigned char* set_valid = (const unsigned char*)d_in[2];
  const float* gene_embed  = (const float*)d_in[3];
  const float* proj_w      = (const float*)d_in[4];
  const float* proj_b      = (const float*)d_in[5];
  const float* proj_ln_g   = (const float*)d_in[6];
  const float* proj_ln_b   = (const float*)d_in[7];
  const float* in_proj_w   = (const float*)d_in[8];
  const float* in_proj_b   = (const float*)d_in[9];
  const float* out_proj_w  = (const float*)d_in[10];
  const float* out_proj_b  = (const float*)d_in[11];
  const float* ln2_g       = (const float*)d_in[12];
  const float* ln2_b       = (const float*)d_in[13];
  float* out = (float*)d_out;

  char* ws = (char*)d_ws;
  __bf16* xbf    = (__bf16*)ws;                                  // 64 MB
  __bf16* projwb = (__bf16*)(ws + (size_t)NB * NG * ND * 2);     // 128x128
  __bf16* inwb   = projwb + ND * ND;                             // 384x128
  __bf16* outwb  = inwb + 3 * ND * ND;                           // 128x128

  gse_cvt<<<(ND * ND + 255) / 256, 256, 0, stream>>>(proj_w, projwb, ND * ND);
  gse_cvt<<<(3 * ND * ND + 255) / 256, 256, 0, stream>>>(in_proj_w, inwb, 3 * ND * ND);
  gse_cvt<<<(ND * ND + 255) / 256, 256, 0, stream>>>(out_proj_w, outwb, ND * ND);

  gse_proj<<<(NB * NG) / 128, 256, 0, stream>>>(
      gene_expr, gene_embed, projwb, proj_b, proj_ln_g, proj_ln_b, xbf);

  gse_attn<<<dim3(NB, NS), 256, 0, stream>>>(
      xbf, set_indices, set_valid, inwb, in_proj_b, outwb, out_proj_b,
      ln2_g, ln2_b, out);
}
